// MultiHeadSelfAttention_33243046871680
// MI455X (gfx1250) — compile-verified, matrix operand order checked by reading, not run on hardware
//
#include <hip/hip_runtime.h>
#include <hip/hip_bf16.h>
#include <stdint.h>

// ---------------------------------------------------------------------------
// Multi-head self-attention with RoPE for MI455X (gfx1250, wave32, WMMA).
// All matmuls (Q/K/V/O projections, Q*K^T, P*V, softmax row-sums) run on
// v_wmma_f32_16x16x32_f16 with f32 accumulation; softmax max-reduction uses
// packed-f16 butterflies (v_pk_max_num_f16) to halve DS-permute traffic.
// ---------------------------------------------------------------------------

#define BATCH   2
#define SEQ     2048
#define DMODEL  1024
#define NHEADS  16
#define DK      64
#define MROWS   (BATCH * SEQ)      // 4096

typedef __attribute__((ext_vector_type(16))) _Float16 v16h;
typedef __attribute__((ext_vector_type(8)))  float    v8f;
typedef __attribute__((ext_vector_type(2)))  _Float16 h2;

union H2I { h2 h; int i; };

// ---------------------------------------------------------------------------
// Fragment loader for 16-bit A/B matrices (CDNA5 ISA 16-bit 16x32 layout):
// lane L: row = L%16, half = L/16.
//   VGPR v in 0..3 : K = k0 + half*8 + 2v     (pair)
//   VGPR v in 4..7 : K = k0 + half*8 + 16+2(v-4)
// Works for:
//   A from row-major [M,K]             (row0 = m tile base)
//   B = W^T where W is row-major [N,K] (row0 = n tile base) -- contiguous pairs
// ---------------------------------------------------------------------------
__device__ __forceinline__ v16h load_frag16(const _Float16* __restrict__ p,
                                            int ld, int row0, int k0, int lane) {
  const int r    = lane & 15;
  const int half = lane >> 4;
  const _Float16* base = p + (size_t)(row0 + r) * ld + k0 + half * 8;
  union { v16h v; unsigned u[8]; } f;
#pragma unroll
  for (int i = 0; i < 4; ++i)
    f.u[i] = *reinterpret_cast<const unsigned*>(base + 2 * i);
#pragma unroll
  for (int i = 0; i < 4; ++i)
    f.u[4 + i] = *reinterpret_cast<const unsigned*>(base + 16 + 2 * i);
  return f.v;
}

__device__ __forceinline__ v8f wmma_f16(v16h a, v16h b, v8f c) {
  // D = A(16x32 f16) * B(32x16 f16) + C(16x16 f32)
  return __builtin_amdgcn_wmma_f32_16x16x32_f16(false, a, false, b,
                                                (short)0, c, false, false);
}

// ---------------------------------------------------------------------------
// f32 -> f16 conversion
// ---------------------------------------------------------------------------
__global__ void cvt_f32_to_f16(const float* __restrict__ in,
                               _Float16* __restrict__ out, int n) {
  int i = blockIdx.x * blockDim.x + threadIdx.x;
  if (i < n) out[i] = (_Float16)in[i];
}

// ---------------------------------------------------------------------------
// C[M,N] = A[M,K] (f16, row-major) * W[N,K]^T (f16, row-major)
// Block = 256 threads = 8 waves; wave tile 32x64 (8 WMMA : 6 frag loads);
// block tile 128(M) x 128(N).
// ---------------------------------------------------------------------------
template <bool HALF_OUT>
__global__ void __launch_bounds__(256)
gemm_xwT(const _Float16* __restrict__ A, const _Float16* __restrict__ W,
         float* __restrict__ Cf, _Float16* __restrict__ Ch,
         int M, int N, int K) {
  const int lane = threadIdx.x & 31;
  const int wave = threadIdx.x >> 5;
  const int m0 = blockIdx.y * 128 + (wave & 3) * 32;   // 4 M-tiles of 32
  const int n0 = blockIdx.x * 128 + (wave >> 2) * 64;  // 2 N-tiles of 64

  v8f acc[2][4] = {};

  for (int k0 = 0; k0 < K; k0 += 32) {
    if (k0 + 32 < K) {   // CDNA5 global_prefetch for the next k-slab
      __builtin_prefetch(A + (size_t)m0 * K + k0 + 32, 0, 1);
      __builtin_prefetch(W + (size_t)n0 * K + k0 + 32, 0, 1);
    }
    const v16h a0 = load_frag16(A, K, m0,      k0, lane);
    const v16h a1 = load_frag16(A, K, m0 + 16, k0, lane);
#pragma unroll
    for (int j = 0; j < 4; ++j) {
      const v16h b = load_frag16(W, K, n0 + 16 * j, k0, lane);
      acc[0][j] = wmma_f16(a0, b, acc[0][j]);
      acc[1][j] = wmma_f16(a1, b, acc[1][j]);
    }
  }

  // C/D layout: lane col = lane%16; VGPR r holds row r + (lane/16)*8.
  const int col = lane & 15, rbase = (lane >> 4) * 8;
#pragma unroll
  for (int i = 0; i < 2; ++i) {
#pragma unroll
    for (int r = 0; r < 8; ++r) {
      const size_t rowoff = (size_t)(m0 + 16 * i + rbase + r) * N;
#pragma unroll
      for (int j = 0; j < 4; ++j) {
        const size_t idx = rowoff + n0 + 16 * j + col;
        if (HALF_OUT) Ch[idx] = (_Float16)acc[i][j][r];
        else          Cf[idx] = acc[i][j][r];
      }
    }
  }
}

// ---------------------------------------------------------------------------
// RoPE (interleaved pairs) on Q,K + repack:
//   Qtmp/Ktmp/Vtmp: f16 [B*S, DMODEL]  (row b*S+s, col h*64+d)
//   Qh, Kh:         f16 [B,H,S,DK]
//   Vt:             f16 [B,H,DK,S]   (transposed for P*V B-fragments)
// One thread per (b,s,h,pair j).
// ---------------------------------------------------------------------------
__global__ void rope_pack(const _Float16* __restrict__ Qtmp,
                          const _Float16* __restrict__ Ktmp,
                          const _Float16* __restrict__ Vtmp,
                          _Float16* __restrict__ Qh,
                          _Float16* __restrict__ Kh,
                          _Float16* __restrict__ Vt) {
  int idx = blockIdx.x * blockDim.x + threadIdx.x;   // < B*S*H*32
  const int j = idx & 31;            // pair index, d = 2j, 2j+1
  const int h = (idx >> 5) & (NHEADS - 1);
  const int s = (idx >> 9) & (SEQ - 1);
  const int b = idx >> 20;
  if (b >= BATCH) return;

  const size_t row = (size_t)(b * SEQ + s) * DMODEL + h * DK + 2 * j;
  const float inv_freq = __expf(-((float)(2 * j) / (float)DK) * 9.210340372f); // ln(1e4)
  const float ang = (float)s * inv_freq;
  float c, sn;
  __sincosf(ang, &sn, &c);

  const float q0 = (float)Qtmp[row], q1 = (float)Qtmp[row + 1];
  const float k0 = (float)Ktmp[row], k1 = (float)Ktmp[row + 1];

  const size_t hd  = (size_t)(b * NHEADS + h);
  const size_t qko = (hd * SEQ + s) * DK + 2 * j;
  Qh[qko]     = (_Float16)(q0 * c - q1 * sn);
  Qh[qko + 1] = (_Float16)(q1 * c + q0 * sn);
  Kh[qko]     = (_Float16)(k0 * c - k1 * sn);
  Kh[qko + 1] = (_Float16)(k1 * c + k0 * sn);

  // V transpose: Vt[b,h,d,s]
  const size_t vto = (hd * DK + 2 * j) * SEQ + s;
  Vt[vto]       = Vtmp[row];
  Vt[vto + SEQ] = Vtmp[row + 1];
}

// ---------------------------------------------------------------------------
// One 32-key flash-attention step for a 16-row Q tile.
// MASKED=false: block fully below the causal diagonal -> no compare/select.
// Row-max: packed-f16 butterflies (softmax only needs an upper-bound-quality
// max, so f16 rounding is harmless). Row-sum: WMMA with an all-ones B.
// ---------------------------------------------------------------------------
template <bool MASKED>
__device__ __forceinline__ void attn_step(
    const _Float16* __restrict__ Kb, const _Float16* __restrict__ Vb,
    _Float16* __restrict__ P, v16h aq0, v16h aq1, v16h ones,
    v8f& o0, v8f& o1, v8f& o2, v8f& o3,
    float (&mrow)[8], float (&lrow)[8],
    int key0, int q0, int lane) {
  const int col = lane & 15, rbase = (lane >> 4) * 8;

  // ---- scores S[16 x 32] = Q * K^T, scaled ----
  v8f s0 = {}, s1 = {};
  s0 = wmma_f16(aq0, load_frag16(Kb, DK, key0,      0,  lane), s0);
  s0 = wmma_f16(aq1, load_frag16(Kb, DK, key0,      32, lane), s0);
  s1 = wmma_f16(aq0, load_frag16(Kb, DK, key0 + 16, 0,  lane), s1);
  s1 = wmma_f16(aq1, load_frag16(Kb, DK, key0 + 16, 32, lane), s1);

#pragma unroll
  for (int r = 0; r < 8; ++r) {
    float v0 = s0[r] * 0.125f;                 // 1/sqrt(64)
    float v1 = s1[r] * 0.125f;
    if (MASKED) {
      const int q = q0 + rbase + r;
      if (key0 + col > q)      v0 = -3.0e38f;  // causal mask (-> -inf in f16)
      if (key0 + 16 + col > q) v1 = -3.0e38f;
    }
    s0[r] = v0; s1[r] = v1;
  }

  // ---- per-row block max: 8 rows packed into 4 f16x2, 4 butterfly stages ----
  H2I hm[4];
#pragma unroll
  for (int t = 0; t < 4; ++t) {
    hm[t].h[0] = (_Float16)fmaxf(s0[2 * t],     s1[2 * t]);
    hm[t].h[1] = (_Float16)fmaxf(s0[2 * t + 1], s1[2 * t + 1]);
  }
#pragma unroll
  for (int off = 8; off; off >>= 1) {
#pragma unroll
    for (int t = 0; t < 4; ++t) {
      H2I w;
      w.i = __shfl_xor(hm[t].i, off, 16);
      hm[t].h = __builtin_elementwise_max(hm[t].h, w.h);  // v_pk_max_num_f16
    }
  }

  float alpha[8];
#pragma unroll
  for (int t = 0; t < 4; ++t) {
    const float mlo = fmaxf(mrow[2 * t],     (float)hm[t].h[0]);
    const float mhi = fmaxf(mrow[2 * t + 1], (float)hm[t].h[1]);
    alpha[2 * t]     = __expf(mrow[2 * t]     - mlo);
    alpha[2 * t + 1] = __expf(mrow[2 * t + 1] - mhi);
    mrow[2 * t]     = mlo;
    mrow[2 * t + 1] = mhi;
  }
#pragma unroll
  for (int r = 0; r < 8; ++r) {
    s0[r] = __expf(s0[r] - mrow[r]);
    s1[r] = __expf(s1[r] - mrow[r]);
  }

  // ---- stage P (C/D layout) into LDS row-major, reload as A-fragment ----
#pragma unroll
  for (int r = 0; r < 8; ++r) {
    const int row = rbase + r;
    P[row * 32 + col]      = (_Float16)s0[r];
    P[row * 32 + 16 + col] = (_Float16)s1[r];
  }
  asm volatile("s_wait_dscnt 0" ::: "memory");   // wave-internal LDS RAW
  const v16h ap = load_frag16(P, 32, 0, 0, lane);

  // ---- row sums of (f16-rounded) P via WMMA with all-ones B ----
  v8f sums = {};
  sums = wmma_f16(ap, ones, sums);

  // ---- rescale accumulators/stats, then O += P * V ----
#pragma unroll
  for (int r = 0; r < 8; ++r) {
    o0[r] *= alpha[r]; o1[r] *= alpha[r];
    o2[r] *= alpha[r]; o3[r] *= alpha[r];
    lrow[r] = lrow[r] * alpha[r] + sums[r];
  }
  o0 = wmma_f16(ap, load_frag16(Vb, SEQ, 0,  key0, lane), o0);
  o1 = wmma_f16(ap, load_frag16(Vb, SEQ, 16, key0, lane), o1);
  o2 = wmma_f16(ap, load_frag16(Vb, SEQ, 32, key0, lane), o2);
  o3 = wmma_f16(ap, load_frag16(Vb, SEQ, 48, key0, lane), o3);
}

// ---------------------------------------------------------------------------
// Flash attention, causal. One wave per 16-row Q tile, 4 waves per block.
// Full (unmasked) key blocks first, then the 1-2 diagonal blocks with masking.
// Output: f16 [B*S, DMODEL] (heads re-interleaved) feeding the O projection.
// ---------------------------------------------------------------------------
__global__ void __launch_bounds__(128)
attn_flash(const _Float16* __restrict__ Qh, const _Float16* __restrict__ Kh,
           const _Float16* __restrict__ Vt, _Float16* __restrict__ Oh) {
  __shared__ _Float16 pbuf[4][16 * 32];

  const int lane = threadIdx.x & 31;
  const int wave = threadIdx.x >> 5;
  const int tile = blockIdx.x * 4 + wave;      // < B*H*(S/16) = 4096
  const int qt = tile & 127;
  const int h  = (tile >> 7) & (NHEADS - 1);
  const int b  = tile >> 11;
  const int q0 = qt * 16;

  const size_t hd = (size_t)(b * NHEADS + h);
  const _Float16* Qb = Qh + hd * SEQ * DK;
  const _Float16* Kb = Kh + hd * SEQ * DK;
  const _Float16* Vb = Vt + hd * DK * SEQ;

  const v16h aq0 = load_frag16(Qb, DK, q0, 0,  lane);
  const v16h aq1 = load_frag16(Qb, DK, q0, 32, lane);

  v16h ones;
#pragma unroll
  for (int i = 0; i < 16; ++i) ones[i] = (_Float16)1.0f;

  v8f o0 = {}, o1 = {}, o2 = {}, o3 = {};
  float mrow[8], lrow[8];
#pragma unroll
  for (int r = 0; r < 8; ++r) { mrow[r] = -3.0e38f; lrow[r] = 0.0f; }

  _Float16* P = &pbuf[wave][0];

  // Blocks with key0+31 <= q0 need no causal masking.
  const int fullEnd = (q0 >= 31) ? ((((q0 - 31) >> 5) << 5) + 32) : 0;

  for (int key0 = 0; key0 < fullEnd; key0 += 32)
    attn_step<false>(Kb, Vb, P, aq0, aq1, ones, o0, o1, o2, o3,
                     mrow, lrow, key0, q0, lane);
  for (int key0 = fullEnd; key0 < q0 + 16; key0 += 32)
    attn_step<true>(Kb, Vb, P, aq0, aq1, ones, o0, o1, o2, o3,
                    mrow, lrow, key0, q0, lane);

  // ---- normalize and emit [B*S, DMODEL] f16 for the O projection ----
  const int col = lane & 15, rbase = (lane >> 4) * 8;
#pragma unroll
  for (int r = 0; r < 8; ++r) {
    const int q = q0 + rbase + r;
    const size_t ro = (size_t)(b * SEQ + q) * DMODEL + h * DK;
    const float inv = 1.0f / lrow[r];
    Oh[ro + col]      = (_Float16)(o0[r] * inv);
    Oh[ro + 16 + col] = (_Float16)(o1[r] * inv);
    Oh[ro + 32 + col] = (_Float16)(o2[r] * inv);
    Oh[ro + 48 + col] = (_Float16)(o3[r] * inv);
  }
}

// ---------------------------------------------------------------------------
// Host launcher
// ---------------------------------------------------------------------------
extern "C" void kernel_launch(void* const* d_in, const int* in_sizes, int n_in,
                              void* d_out, int out_size, void* d_ws, size_t ws_size,
                              hipStream_t stream) {
  (void)in_sizes; (void)n_in; (void)out_size; (void)ws_size;

  const float* x  = (const float*)d_in[0];
  const float* wq = (const float*)d_in[1];
  const float* wk = (const float*)d_in[2];
  const float* wv = (const float*)d_in[3];
  const float* wo = (const float*)d_in[4];
  float* out = (float*)d_out;

  char* ws = (char*)d_ws;
  auto carve = [&](size_t bytes) {
    char* p = ws;
    ws += (bytes + 255) & ~(size_t)255;
    return p;
  };
  const size_t actBytes = (size_t)MROWS * DMODEL * sizeof(_Float16);   // 8 MB
  const size_t wBytes   = (size_t)DMODEL * DMODEL * sizeof(_Float16);  // 2 MB

  _Float16* xh      = (_Float16*)carve(actBytes);
  _Float16* wqh     = (_Float16*)carve(wBytes);
  _Float16* wkh     = (_Float16*)carve(wBytes);
  _Float16* wvh     = (_Float16*)carve(wBytes);
  _Float16* woh     = (_Float16*)carve(wBytes);
  _Float16* qtmp    = (_Float16*)carve(actBytes);
  _Float16* ktmp    = (_Float16*)carve(actBytes);
  _Float16* vtmp    = (_Float16*)carve(actBytes);
  _Float16* Qh      = (_Float16*)carve(actBytes);   // [B,H,S,DK]
  _Float16* Kh      = (_Float16*)carve(actBytes);
  _Float16* Vt      = (_Float16*)carve(actBytes);   // [B,H,DK,S]
  _Float16* attnOut = (_Float16*)carve(actBytes);

  // 1) f32 -> f16
  {
    const int nAct = MROWS * DMODEL, nW = DMODEL * DMODEL;
    cvt_f32_to_f16<<<(nAct + 255) / 256, 256, 0, stream>>>(x,  xh,  nAct);
    cvt_f32_to_f16<<<(nW + 255) / 256, 256, 0, stream>>>(wq, wqh, nW);
    cvt_f32_to_f16<<<(nW + 255) / 256, 256, 0, stream>>>(wk, wkh, nW);
    cvt_f32_to_f16<<<(nW + 255) / 256, 256, 0, stream>>>(wv, wvh, nW);
    cvt_f32_to_f16<<<(nW + 255) / 256, 256, 0, stream>>>(wo, woh, nW);
  }

  // 2) Q/K/V projections (f16 out); block tile 128x128
  dim3 gGemm(DMODEL / 128, MROWS / 128);
  gemm_xwT<true><<<gGemm, 256, 0, stream>>>(xh, wqh, nullptr, qtmp,
                                            MROWS, DMODEL, DMODEL);
  gemm_xwT<true><<<gGemm, 256, 0, stream>>>(xh, wkh, nullptr, ktmp,
                                            MROWS, DMODEL, DMODEL);
  gemm_xwT<true><<<gGemm, 256, 0, stream>>>(xh, wvh, nullptr, vtmp,
                                            MROWS, DMODEL, DMODEL);

  // 3) RoPE + repack (Q,K -> [B,H,S,DK]; V -> [B,H,DK,S])
  {
    const int n = BATCH * SEQ * NHEADS * 32;
    rope_pack<<<n / 256, 256, 0, stream>>>(qtmp, ktmp, vtmp, Qh, Kh, Vt);
  }

  // 4) causal flash attention: 4096 q-tiles, 4 waves/block
  attn_flash<<<(BATCH * NHEADS * (SEQ / 16)) / 4, 128, 0, stream>>>(
      Qh, Kh, Vt, attnOut);

  // 5) output projection (f32 out)
  gemm_xwT<false><<<gGemm, 256, 0, stream>>>(attnOut, woh, out, nullptr,
                                             MROWS, DMODEL, DMODEL);
}
